// SNNClassifier_77129022702058
// MI455X (gfx1250) — matrix operand order repaired and checked by reading, not compiled, not run on hardware
//
#include <hip/hip_runtime.h>

typedef __attribute__((ext_vector_type(16))) __bf16 v16bf;
typedef __attribute__((ext_vector_type(8)))  __bf16 v8bf;
typedef __attribute__((ext_vector_type(8)))  float  v8f;
typedef __attribute__((ext_vector_type(4)))  float  f32x4;
typedef __attribute__((ext_vector_type(4)))  unsigned int u32x4;
typedef __attribute__((ext_vector_type(2)))  unsigned int u32x2;
typedef __attribute__((ext_vector_type(8)))  int i32x8;
typedef __attribute__((ext_vector_type(4)))  int i32x4;

namespace {
constexpr int kD     = 256;   // input dim
constexpr int kH     = 1024;  // hidden dim
constexpr int kC     = 10;    // classes
constexpr int kTileM = 128;   // rows per block (8 waves x 16)
constexpr int kXTStr = 264;   // LDS halfs per x row (16B aligned, bank-skewed)
constexpr int kWTStr = 72;    // LDS halfs per W1t row chunk (64 + 8 pad = TDM pad 4 dwords/32 dwords)
constexpr int kRTStr = 72;    // LDS halfs per rate row
}

static __device__ __forceinline__ unsigned short f2bf(float f) {
    unsigned int u = __builtin_bit_cast(unsigned int, f);
    u += 0x7FFFu + ((u >> 16) & 1u);          // round-to-nearest-even
    return (unsigned short)(u >> 16);
}

// build a 16-elem bf16 fragment from two 16B groups (8 halfs each)
static __device__ __forceinline__ v16bf mk_frag(const unsigned short* lo,
                                                const unsigned short* hi) {
    v8bf a = __builtin_bit_cast(v8bf, *(const u32x4*)lo);
    v8bf b = __builtin_bit_cast(v8bf, *(const u32x4*)hi);
    return __builtin_shufflevector(a, b, 0,1,2,3,4,5,6,7,8,9,10,11,12,13,14,15);
}

// ---- TDM: DMA one W1t chunk [256 rows x 64 cols bf16, row stride 1024] into
//      LDS with hardware padding of 4 DWORDs after every 32 DWORDs
//      (-> 72-half LDS row stride). Issued once per wave; EXEC-independent. ----
static __device__ __forceinline__ void tdm_load_w1t_chunk(const unsigned short* gsrc,
                                                          unsigned lds_off) {
    unsigned long long ga = (unsigned long long)(size_t)gsrc;
    u32x4 g0;
    g0.x = 1u;                                   // count=1, user mode
    g0.y = lds_off;                              // lds_addr (bytes)
    g0.z = (unsigned)ga;                         // global_addr[31:0]
    g0.w = (unsigned)(ga >> 32) | (2u << 30);    // global_addr[56:32] | type=2
    i32x8 g1;
    g1[0] = (int)((1u << 16)    // data_size = 2B
                | (1u << 20)    // pad_enable
                | (4u << 22)    // pad_interval: every 32 DWORDs (128B)
                | (3u << 25));  // pad_amount: 4 DWORDs (16B)
    g1[1] = (int)(1024u << 16); // tensor_dim0 = 1024 (lo16)
    g1[2] = (int)(256u << 16);  // tensor_dim0 hi | tensor_dim1 = 256 (lo16)
    g1[3] = (int)(64u << 16);   // tensor_dim1 hi | tile_dim0 = 64
    g1[4] = 256;                // tile_dim1 = 256 | tile_dim2 = 0
    g1[5] = 1024;               // tensor_dim0_stride = 1024 (lo32)
    g1[6] = 0;                  // stride hi | tensor_dim1_stride lo
    g1[7] = 0;
    i32x4 gz = {0, 0, 0, 0};
#if __clang_major__ >= 23
    i32x8 gz8 = {0, 0, 0, 0, 0, 0, 0, 0};
    __builtin_amdgcn_tensor_load_to_lds(g0, g1, gz, gz, gz8, 0);
#else
    __builtin_amdgcn_tensor_load_to_lds(g0, g1, gz, gz, 0);
#endif
}

// ---- pre-pass: W1 [1024x256] f32 -> W1t [256x1024] bf16 (K-major) ----
__global__ __launch_bounds__(256) void prep_w1t(const float* __restrict__ W1,
                                                unsigned short* __restrict__ W1t) {
    const int idx = blockIdx.x * 256 + threadIdx.x;   // 0..262143
    const int k = idx & (kD - 1);
    const int n = idx >> 8;
    W1t[k * kH + n] = f2bf(W1[n * kD + k]);
}

// ---- pre-pass: W2 [10x1024] f32 -> W2p [1024x16] bf16, cols 10..15 zero ----
__global__ __launch_bounds__(256) void prep_w2p(const float* __restrict__ W2,
                                                unsigned short* __restrict__ W2p) {
    const int idx = blockIdx.x * 256 + threadIdx.x;   // 0..16383
    const int n = idx & 15;
    const int h = idx >> 4;
    W2p[idx] = (n < kC) ? f2bf(W2[n * kH + h]) : (unsigned short)0;
}

// ---- fused: fc1 (WMMA bf16, TDM-fed) -> closed-form LIF rate -> fc2 (WMMA bf16) ----
__global__ __launch_bounds__(256, 1) void snn_fused(
    const float* __restrict__ x,              // [32768][256]
    const float* __restrict__ b1,             // [1024]
    const float* __restrict__ b2,             // [10]
    const unsigned short* __restrict__ W1t,   // [256][1024] bf16
    const unsigned short* __restrict__ W2p,   // [1024][16]  bf16
    float* __restrict__ out)                  // [32768][10]
{
    __shared__ unsigned short XT[kTileM * kXTStr];     // 67584 B  x tile (bf16)
    __shared__ unsigned short WT[2][kD * kWTStr];      // 2x36864 B W1t chunks (TDM double buffer)
    __shared__ unsigned short RT[8 * 16 * kRTStr];     // 18432 B  per-wave rate staging

    const int tid  = threadIdx.x;
    const int wave = tid >> 5;
    const int lane = tid & 31;
    const int m0   = blockIdx.x * kTileM;

    // kick off DMA of chunk 0 while we convert the x tile
    if (wave == 0)
        tdm_load_w1t_chunk(W1t, (unsigned)(size_t)&WT[0][0]);

    // ---- stage x tile -> bf16 in LDS (coalesced f32x4 loads) ----
    {
        const int sub = tid >> 6;          // row subgroup 0..3
        const int c4  = (tid & 63) << 2;   // float column, multiple of 4
        #pragma unroll
        for (int i = 0; i < 32; ++i) {
            const int row = i * 4 + sub;
            f32x4 v = *(const f32x4*)(x + (size_t)(m0 + row) * kD + c4);
            u32x2 p;
            p.x = (unsigned int)f2bf(v.x) | ((unsigned int)f2bf(v.y) << 16);
            p.y = (unsigned int)f2bf(v.z) | ((unsigned int)f2bf(v.w) << 16);
            *(u32x2*)(XT + row * kXTStr + c4) = p;
        }
    }
    __syncthreads();

    const int mrow  = wave << 4;         // wave's 16 rows in tile
    const int ncol  = lane & 15;         // N column within 16 (C layout)
    const int rbase = (lane >> 4) << 3;  // 0 or 8 (C layout M offset)
    const v8f vzero = {0.f,0.f,0.f,0.f,0.f,0.f,0.f,0.f};
    v8f outacc = vzero;                  // fc2 accumulator 16x16 (10 valid cols)

    for (int nc = 0; nc < kH / 64; ++nc) {
        const int n0  = nc * 64;
        const int buf = nc & 1;

        // chunk nc has landed; release all waves, then prefetch chunk nc+1
        if (wave == 0)
            __builtin_amdgcn_s_wait_tensorcnt(0);
        __syncthreads();
        if (wave == 0 && nc + 1 < kH / 64)
            tdm_load_w1t_chunk(W1t + n0 + 64, (unsigned)(size_t)&WT[buf ^ 1][0]);

        // ---- fc1: 16x64 tile, K=256, via 32 WMMA bf16 ----
        v8f acc[4];
        #pragma unroll
        for (int j = 0; j < 4; ++j) acc[j] = vzero;

        const unsigned short* arowp =
            XT + (mrow + ncol) * kXTStr + ((lane >> 4) << 3);
        const unsigned short* wtb = &WT[buf][0];
        #pragma unroll
        for (int kt = 0; kt < 8; ++kt) {
            const int kb = kt * 32;
            v16bf A = mk_frag(arowp + kb, arowp + kb + 16);      // A 16x32 layout
            const unsigned short* brow = wtb + (kb + ((lane >> 4) << 3)) * kWTStr + ncol; // B: lane = N col, elems over K
            #pragma unroll
            for (int j = 0; j < 4; ++j) {
                v16bf Bf;
                #pragma unroll
                for (int s = 0; s < 8; ++s) {
                    Bf[s]     = __builtin_bit_cast(__bf16, brow[s * kWTStr + j * 16]);
                    Bf[s + 8] = __builtin_bit_cast(__bf16, brow[(16 + s) * kWTStr + j * 16]);
                }
                acc[j] = __builtin_amdgcn_wmma_f32_16x16x32_bf16(
                    false, A, false, Bf, (short)0, acc[j], false, false);
            }
        }

        // ---- LIF closed form: with constant input h the dynamics are periodic:
        //      first spike at k* = ceil(log2(h/(h-1))) (h>1), count = floor(16/k*) ----
        unsigned short* rtw = RT + wave * (16 * kRTStr);
        #pragma unroll
        for (int j = 0; j < 4; ++j) {
            const float b1v = b1[n0 + j * 16 + ncol];  // one N col per lane (C layout)
            #pragma unroll
            for (int r = 0; r < 8; ++r) {
                const float h = acc[j][r] + b1v;
                const float d = h - 1.0f;
                const float k = __builtin_ceilf(
                    __builtin_amdgcn_logf(h * __builtin_amdgcn_rcpf(d)));
                float cnt = (d > 0.0f && k <= 16.0f)
                          ? __builtin_floorf(16.0f * __builtin_amdgcn_rcpf(k))
                          : 0.0f;
                rtw[(r + rbase) * kRTStr + j * 16 + ncol] = f2bf(cnt * 0.0625f);
            }
        }
        // RT is produced and consumed by the same wave; DS ops are in-order per wave.

        // ---- fc2 partial: out += rate(16x64) @ W2p(64x16), 2 WMMA ----
        {
            const unsigned short* rrow =
                rtw + ncol * kRTStr + ((lane >> 4) << 3);
            #pragma unroll
            for (int ks = 0; ks < 2; ++ks) {
                v16bf A2 = mk_frag(rrow + ks * 32, rrow + ks * 32 + 16);
                const unsigned short* wp = W2p + (size_t)(n0 + ks * 32 + ((lane >> 4) << 3)) * 16 + ncol;
                v16bf B2;
                #pragma unroll
                for (int s = 0; s < 8; ++s) {
                    B2[s]     = __builtin_bit_cast(__bf16, wp[s * 16]);
                    B2[s + 8] = __builtin_bit_cast(__bf16, wp[(16 + s) * 16]);
                }
                outacc = __builtin_amdgcn_wmma_f32_16x16x32_bf16(
                    false, A2, false, B2, (short)0, outacc, false, false);
            }
        }
        __syncthreads();   // all waves done with WT[buf] before TDM reuses it
    }

    // ---- epilogue: out = outacc + b2 (C layout store) ----
    if (ncol < kC) {
        const float b2v = b2[ncol];
        #pragma unroll
        for (int r = 0; r < 8; ++r) {
            const int row = m0 + mrow + rbase + r;
            out[(size_t)row * kC + ncol] = outacc[r] + b2v;
        }
    }
}

extern "C" void kernel_launch(void* const* d_in, const int* in_sizes, int n_in,
                              void* d_out, int out_size, void* d_ws, size_t ws_size,
                              hipStream_t stream) {
    const float* x  = (const float*)d_in[0];
    const float* W1 = (const float*)d_in[1];
    const float* b1 = (const float*)d_in[2];
    const float* W2 = (const float*)d_in[3];
    const float* b2 = (const float*)d_in[4];
    (void)n_in; (void)out_size; (void)ws_size;
    float* out = (float*)d_out;

    unsigned short* W1t = (unsigned short*)d_ws;            // 512 KB
    unsigned short* W2p = W1t + (size_t)kD * kH;            // +32 KB

    prep_w1t<<<(kD * kH) / 256, 256, 0, stream>>>(W1, W1t);
    prep_w2p<<<(kH * 16) / 256, 256, 0, stream>>>(W2, W2p);

    const int B = in_sizes[0] / kD;                         // 32768
    snn_fused<<<B / kTileM, 256, 0, stream>>>(x, b1, b2, W1t, W2p, out);
}
